// DNet_10007273800199
// MI455X (gfx1250) — compile-verified, matrix operand order checked by reading, not run on hardware
//
#include <hip/hip_runtime.h>
#include <hip/hip_bf16.h>

typedef __attribute__((ext_vector_type(16))) __bf16 v16bf;
typedef __attribute__((ext_vector_type(8)))  __bf16 v8bf;
typedef __attribute__((ext_vector_type(4)))  __bf16 v4bf;
typedef __attribute__((ext_vector_type(8)))  float  v8f;
typedef __attribute__((ext_vector_type(4)))  float  v4f;

union ABf { v16bf v; v8bf h[2]; };

__device__ __forceinline__ float leakyf(float x, float s) { return x > 0.f ? x : s * x; }
// monotonic float<->uint key for atomicMax-based segment max
__device__ __forceinline__ unsigned fkey(float f) {
    unsigned u = __float_as_uint(f);
    return (u & 0x80000000u) ? ~u : (u | 0x80000000u);
}
__device__ __forceinline__ float unfkey(unsigned k) {
    return __uint_as_float((k & 0x80000000u) ? (k ^ 0x80000000u) : ~k);
}

// ---------------- utility kernels ----------------
__global__ void k_f32_to_bf16(const float* __restrict__ src, __bf16* __restrict__ dst, int n) {
    int i = blockIdx.x * blockDim.x + threadIdx.x;
    if (i < n) dst[i] = (__bf16)src[i];
}
__global__ void k_fill_u32(unsigned* __restrict__ p, unsigned v, int n) {
    int i = blockIdx.x * blockDim.x + threadIdx.x;
    if (i < n) p[i] = v;
}
__global__ void k_fill_f32(float* __restrict__ p, float v, int n) {
    int i = blockIdx.x * blockDim.x + threadIdx.x;
    if (i < n) p[i] = v;
}

// ---------------- WMMA GEMM: D[M,1024] = A[M,128] * W[1024,128]^T + bias ----------------
// One wave computes a 16x64 tile (4 accumulators sharing one A fragment).
// Per k-step: issue ALL loads (A + 4xB) first, then 4 WMMAs -> staggered loadcnt waits,
// loads overlap compute. K=128 hardcoded -> fully unrolled, immediate address offsets.
__global__ __launch_bounds__(256) void k_gemm_bf16(
    const __bf16* __restrict__ A, const __bf16* __restrict__ W,
    const float* __restrict__ bias, __bf16* __restrict__ D,
    int M, int Nn)
{
    constexpr int K = 128;
    int wave = (blockIdx.x * blockDim.x + threadIdx.x) >> 5;
    int lane = threadIdx.x & 31;
    int tilesN4 = Nn >> 6;                 // 16x64 tiles along N
    int tm = wave / tilesN4, tn4 = wave % tilesN4;
    if (tm >= (M >> 4)) return;
    int half = lane >> 4, l16 = lane & 15;
    int rowA = (tm << 4) + l16;
    int koff = half << 3;                  // lanes 0-15: K 0..7/16..23 ; lanes 16-31: K 8..15/24..31
    const __bf16* arow  = A + (size_t)rowA * K + koff;
    const __bf16* brow0 = W + (size_t)((tn4 << 6) + l16) * K + koff;  // col j*16+l16 -> +j*16*K
    v8f acc[4] = {v8f{}, v8f{}, v8f{}, v8f{}};
    #pragma unroll
    for (int k0 = 0; k0 < K; k0 += 32) {
        ABf a, b[4];
        a.h[0] = *(const v8bf*)(arow + k0);
        a.h[1] = *(const v8bf*)(arow + k0 + 16);
        #pragma unroll
        for (int j = 0; j < 4; ++j) {
            const __bf16* brow = brow0 + j * (16 * K);
            b[j].h[0] = *(const v8bf*)(brow + k0);
            b[j].h[1] = *(const v8bf*)(brow + k0 + 16);
        }
        #pragma unroll
        for (int j = 0; j < 4; ++j)
            acc[j] = __builtin_amdgcn_wmma_f32_16x16x32_bf16(false, a.v, false, b[j].v,
                                                             (short)0, acc[j], false, false);
    }
    #pragma unroll
    for (int j = 0; j < 4; ++j) {
        int colB = (tn4 << 6) + (j << 4) + l16;
        float bv = bias ? bias[colB] : 0.f;
        #pragma unroll
        for (int r = 0; r < 8; ++r) {
            int row = (tm << 4) + r + (half << 3);  // C layout: vgpr r, lane<16 -> M=r, lane>=16 -> M=r+8
            D[(size_t)row * Nn + colB] = (__bf16)(acc[j][r] + bv);
        }
    }
}

// ---------------- attention score: one wave per (edge, head) ----------------
__global__ __launch_bounds__(256) void k_edge_score(
    const __bf16* __restrict__ XL, const __bf16* __restrict__ XR,
    const long long* __restrict__ si, const long long* __restrict__ di,
    const float* __restrict__ att, float* __restrict__ s,
    unsigned* __restrict__ smaxKey, int E, int Etot)
{
    int wave = (blockIdx.x * blockDim.x + threadIdx.x) >> 5;
    int lane = threadIdx.x & 31;
    int e = wave >> 3, h = wave & 7;
    if (e >= Etot) return;
    int src = (e < E) ? (int)si[e] : (e - E);
    int dst = (e < E) ? (int)di[e] : (e - E);
    int c0 = lane << 2;
    v4bf xlv = *(const v4bf*)(XL + (size_t)src * 1024 + h * 128 + c0);
    v4bf xrv = *(const v4bf*)(XR + (size_t)dst * 1024 + h * 128 + c0);
    v4f  av  = *(const v4f*)(att + h * 128 + c0);
    float p = 0.f;
    #pragma unroll
    for (int i = 0; i < 4; ++i) {
        float m = (float)xlv[i] + (float)xrv[i];
        p += leakyf(m, 0.2f) * av[i];
    }
    #pragma unroll
    for (int off = 16; off >= 1; off >>= 1) p += __shfl_xor(p, off, 32);
    if (lane == 0) {
        s[(size_t)e * 8 + h] = p;
        atomicMax(&smaxKey[dst * 8 + h], fkey(p));
    }
}

// ---------------- softmax exp + denominator ----------------
__global__ void k_edge_exp(
    float* __restrict__ s, const unsigned* __restrict__ smaxKey,
    float* __restrict__ denom, const long long* __restrict__ di, int E, int Etot)
{
    int idx = blockIdx.x * blockDim.x + threadIdx.x;
    if (idx >= Etot * 8) return;
    int e = idx >> 3, h = idx & 7;
    int dst = (e < E) ? (int)di[e] : (e - E);
    float ex = __expf(s[idx] - unfkey(smaxKey[dst * 8 + h]));
    s[idx] = ex;
    atomicAdd(&denom[dst * 8 + h], ex);
}

// ---------------- weighted message scatter: one wave per (edge, head) ----------------
__global__ __launch_bounds__(256) void k_edge_scatter(
    const __bf16* __restrict__ XL, const float* __restrict__ s,
    const float* __restrict__ denom,
    const long long* __restrict__ si, const long long* __restrict__ di,
    float* __restrict__ acc, int E, int Etot)
{
    int wave = (blockIdx.x * blockDim.x + threadIdx.x) >> 5;
    int lane = threadIdx.x & 31;
    int e = wave >> 3, h = wave & 7;
    if (e >= Etot) return;
    int src = (e < E) ? (int)si[e] : (e - E);
    int dst = (e < E) ? (int)di[e] : (e - E);
    float alpha = s[(size_t)e * 8 + h] / denom[dst * 8 + h];
    int c0 = lane << 2;
    v4bf xlv = *(const v4bf*)(XL + (size_t)src * 1024 + h * 128 + c0);
    float* base = acc + (size_t)dst * 1024 + h * 128 + c0;
    #pragma unroll
    for (int i = 0; i < 4; ++i) atomicAdd(base + i, (float)xlv[i] * alpha);
}

// ---------------- head mean + bias (+ optional LeakyReLU) -> bf16 node features ----------------
__global__ void k_head_mean(
    const float* __restrict__ acc, const float* __restrict__ bias,
    __bf16* __restrict__ zb, int N, int doLeaky)
{
    int idx = blockIdx.x * blockDim.x + threadIdx.x;
    if (idx >= N * 128) return;
    int n = idx >> 7, c = idx & 127;
    float v = 0.f;
    #pragma unroll
    for (int h = 0; h < 8; ++h) v += acc[(size_t)n * 1024 + h * 128 + c];
    v = v * 0.125f + bias[c];
    if (doLeaky) v = leakyf(v, 0.01f);
    zb[idx] = (__bf16)v;
}

// ---------------- fused decoder ----------------
// One wave per 16-edge tile computes ALL 8 hidden-dim tiles (gather A once per k-step),
// B loads grouped 4-at-a-time ahead of their WMMAs for load/compute overlap,
// then bias+LeakyReLU+dot(Wd2) in-register, lane reduce, plain store (no atomics).
__global__ __launch_bounds__(256) void k_decoder(
    const __bf16* __restrict__ Z,
    const long long* __restrict__ si, const long long* __restrict__ di,
    const __bf16* __restrict__ Wd1, const float* __restrict__ bd1,
    const float* __restrict__ Wd2, const float* __restrict__ bd2,
    float* __restrict__ out, int E)
{
    int wave = (blockIdx.x * blockDim.x + threadIdx.x) >> 5;
    int lane = threadIdx.x & 31;
    int tm = wave;                         // E/16 edge tiles
    if (tm >= (E >> 4)) return;
    int half = lane >> 4, l16 = lane & 15;
    int erow = (tm << 4) + l16;
    int srow = (int)si[erow], drow = (int)di[erow];
    int koff = half << 3;
    const __bf16* wbase = Wd1 + (size_t)l16 * 256 + koff;   // col j*16+l16 -> + j*16*256
    v8f acc[8] = {v8f{}, v8f{}, v8f{}, v8f{}, v8f{}, v8f{}, v8f{}, v8f{}};
    #pragma unroll
    for (int k0 = 0; k0 < 256; k0 += 32) {
        // concat([z[src], z[dst]]): K<128 from src row, K>=128 from dst row
        const __bf16* abase = (k0 < 128) ? (Z + (size_t)srow * 128 + k0 + koff)
                                         : (Z + (size_t)drow * 128 + (k0 - 128) + koff);
        ABf a;
        a.h[0] = *(const v8bf*)(abase);
        a.h[1] = *(const v8bf*)(abase + 16);
        #pragma unroll
        for (int g = 0; g < 2; ++g) {      // two groups of 4 B-tiles: bounded register pressure
            ABf b[4];
            #pragma unroll
            for (int j = 0; j < 4; ++j) {
                const __bf16* wrow = wbase + (g * 4 + j) * (16 * 256);
                b[j].h[0] = *(const v8bf*)(wrow + k0);
                b[j].h[1] = *(const v8bf*)(wrow + k0 + 16);
            }
            #pragma unroll
            for (int j = 0; j < 4; ++j)
                acc[g * 4 + j] = __builtin_amdgcn_wmma_f32_16x16x32_bf16(
                    false, a.v, false, b[j].v, (short)0, acc[g * 4 + j], false, false);
        }
    }
    float bd2v = bd2[0];
    float b1[8], w2[8];
    #pragma unroll
    for (int j = 0; j < 8; ++j) {
        b1[j] = bd1[(j << 4) + l16];
        w2[j] = Wd2[(j << 4) + l16];
    }
    #pragma unroll
    for (int r = 0; r < 8; ++r) {
        float p = 0.f;
        #pragma unroll
        for (int j = 0; j < 8; ++j) p += leakyf(acc[j][r] + b1[j], 0.01f) * w2[j];
        #pragma unroll
        for (int off = 8; off >= 1; off >>= 1) p += __shfl_xor(p, off, 32);  // within 16-lane half
        if (l16 == 0) {
            int eo = (tm << 4) + r + (half << 3);
            out[eo] = p + bd2v;
        }
    }
}

// ---------------- host orchestration ----------------
extern "C" void kernel_launch(void* const* d_in, const int* in_sizes, int n_in,
                              void* d_out, int out_size, void* d_ws, size_t ws_size,
                              hipStream_t stream)
{
    (void)n_in; (void)out_size; (void)ws_size;
    const float*     x    = (const float*)d_in[0];
    const long long* ei   = (const long long*)d_in[1];
    const float* Wl1 = (const float*)d_in[2];  const float* bl1 = (const float*)d_in[3];
    const float* Wr1 = (const float*)d_in[4];  const float* br1 = (const float*)d_in[5];
    const float* att1 = (const float*)d_in[6]; const float* bias1 = (const float*)d_in[7];
    const float* Wl2 = (const float*)d_in[8];  const float* bl2 = (const float*)d_in[9];
    const float* Wr2 = (const float*)d_in[10]; const float* br2 = (const float*)d_in[11];
    const float* att2 = (const float*)d_in[12]; const float* bias2 = (const float*)d_in[13];
    const float* Wd1 = (const float*)d_in[14]; const float* bd1 = (const float*)d_in[15];
    const float* Wd2 = (const float*)d_in[16]; const float* bd2 = (const float*)d_in[17];
    float* out = (float*)d_out;

    const int N = in_sizes[0] / 128;     // 16384
    const int E = in_sizes[1] / 2;       // 262144
    const int Etot = E + N;              // self-loops appended
    const long long* si = ei;
    const long long* dj = ei + E;

    // workspace layout (bf16 XL/XR: 32 MB each -> both + 64 MB f32 accumulator fit in 192 MB L2)
    char* ws = (char*)d_ws;
    size_t off = 0;
    auto alloc = [&](size_t bytes) -> char* {
        char* p = ws + off; off += (bytes + 255) & ~(size_t)255; return p;
    };
    __bf16* XBF  = (__bf16*)alloc((size_t)N * 128 * 2);   // node features (bf16), reused per stage
    __bf16* WLb  = (__bf16*)alloc((size_t)1024 * 128 * 2);
    __bf16* WRb  = (__bf16*)alloc((size_t)1024 * 128 * 2);
    __bf16* WD1b = (__bf16*)alloc((size_t)128 * 256 * 2);
    __bf16* XL   = (__bf16*)alloc((size_t)N * 1024 * 2);
    __bf16* XR   = (__bf16*)alloc((size_t)N * 1024 * 2);
    float*  ACC  = (float*)alloc((size_t)N * 1024 * 4);
    float*  S    = (float*)alloc((size_t)Etot * 8 * 4);
    unsigned* SMAX = (unsigned*)alloc((size_t)N * 8 * 4);
    float*  DEN  = (float*)alloc((size_t)N * 8 * 4);

    const int T = 256;
    auto cdiv = [](int a, int b) { return (a + b - 1) / b; };

    // x -> bf16
    k_f32_to_bf16<<<cdiv(N * 128, T), T, 0, stream>>>(x, XBF, N * 128);

    auto run_layer = [&](const float* Wl, const float* bl, const float* Wr, const float* br,
                         const float* att, const float* bias, int doLeaky) {
        k_f32_to_bf16<<<cdiv(1024 * 128, T), T, 0, stream>>>(Wl, WLb, 1024 * 128);
        k_f32_to_bf16<<<cdiv(1024 * 128, T), T, 0, stream>>>(Wr, WRb, 1024 * 128);
        int gemmWaves = (N / 16) * (1024 / 64);          // 16x64 tiles
        k_gemm_bf16<<<cdiv(gemmWaves, 8), T, 0, stream>>>(XBF, WLb, bl, XL, N, 1024);
        k_gemm_bf16<<<cdiv(gemmWaves, 8), T, 0, stream>>>(XBF, WRb, br, XR, N, 1024);
        k_fill_u32<<<cdiv(N * 8, T), T, 0, stream>>>(SMAX, 0u, N * 8);
        k_fill_f32<<<cdiv(N * 8, T), T, 0, stream>>>(DEN, 0.f, N * 8);
        k_fill_f32<<<cdiv(N * 1024, T), T, 0, stream>>>(ACC, 0.f, N * 1024);
        int edgeWaves = Etot * 8;
        int edgeBlocks = cdiv(edgeWaves, 8);             // 8 waves / 256-thread block
        k_edge_score<<<edgeBlocks, T, 0, stream>>>(XL, XR, si, dj, att, S, SMAX, E, Etot);
        k_edge_exp<<<cdiv(Etot * 8, T), T, 0, stream>>>(S, SMAX, DEN, dj, E, Etot);
        k_edge_scatter<<<edgeBlocks, T, 0, stream>>>(XL, S, DEN, si, dj, ACC, E, Etot);
        k_head_mean<<<cdiv(N * 128, T), T, 0, stream>>>(ACC, bias, XBF, N, doLeaky);
    };

    run_layer(Wl1, bl1, Wr1, br1, att1, bias1, /*leaky=*/1);
    run_layer(Wl2, bl2, Wr2, br2, att2, bias2, /*leaky=*/0);

    // decoder: one wave per 16 edges, full 128 hidden dims in-register
    k_f32_to_bf16<<<cdiv(128 * 256, T), T, 0, stream>>>(Wd1, WD1b, 128 * 256);
    int decWaves = E / 16;
    k_decoder<<<cdiv(decWaves, 8), T, 0, stream>>>(XBF, si, dj, WD1b, bd1, Wd2, bd2, out, E);
}
